// FrameTokenizer_61692910240284
// MI455X (gfx1250) — matrix operand order repaired and checked by reading, not compiled, not run on hardware
//
#include <hip/hip_runtime.h>
#include <math.h>

// ---------------------------------------------------------------------------
// Problem constants (from reference)
// ---------------------------------------------------------------------------
#define B_ 16
#define S_ 3000
#define T_ 512
#define DENC 1280
#define DHID 1024
#define DEMB 1024
#define M_TOTAL (B_ * T_)       // 8192 rows through the MLP
#define LN_EPS 1e-5f

// ---------------------------------------------------------------------------
// CDNA5 WMMA vector types (wave32)
// ---------------------------------------------------------------------------
typedef __attribute__((ext_vector_type(16))) __bf16 v16bf;
typedef __attribute__((ext_vector_type(8)))  __bf16 v8bf;
typedef __attribute__((ext_vector_type(8)))  float  v8f;

// ---------------------------------------------------------------------------
// Kernel 1: ragged span-mean pooling, fp32 -> bf16
// One block per (b, t) token. 256 threads, each owns 5 columns (5*256=1280).
// Reads x exactly once (coalesced rows), writes pooled bf16 [8192,1280].
// ---------------------------------------------------------------------------
__global__ __launch_bounds__(256) void pool_kernel(
    const float* __restrict__ x,        // [B,S,DENC]
    const int*   __restrict__ ts,       // [B,T,2]
    __bf16*      __restrict__ pooled)   // [B*T, DENC] bf16
{
    const int bid  = blockIdx.x;             // b*T + t
    const int b    = bid / T_;
    const int tid  = threadIdx.x;
    const int start = ts[bid * 2 + 0];
    const int end   = ts[bid * 2 + 1];
    const int len   = end - start;
    const float* xb = x + (size_t)b * S_ * DENC;

    float acc[5] = {0.f, 0.f, 0.f, 0.f, 0.f};
    if (len > 0) {
        for (int s = start; s < end; ++s) {
            const float* row = xb + (size_t)s * DENC + tid;
            #pragma unroll
            for (int i = 0; i < 5; ++i) acc[i] += row[i * 256];
        }
        const float inv = 1.f / (float)len;
        #pragma unroll
        for (int i = 0; i < 5; ++i) acc[i] *= inv;
    } else {
        // empty-span fallback: x[b, 0, :]
        const float* row = xb + tid;
        #pragma unroll
        for (int i = 0; i < 5; ++i) acc[i] = row[i * 256];
    }
    __bf16* dst = pooled + (size_t)bid * DENC + tid;
    #pragma unroll
    for (int i = 0; i < 5; ++i) dst[i * 256] = (__bf16)acc[i];
}

// ---------------------------------------------------------------------------
// Kernel 2: weight transpose + bf16 convert: W [K,N] f32 -> Wt [N,K] bf16
// so B-matrix WMMA fragments become contiguous 32-byte loads.
// ---------------------------------------------------------------------------
__global__ __launch_bounds__(256) void transpose_to_bf16_kernel(
    const float* __restrict__ W, __bf16* __restrict__ Wt, int K, int N)
{
    __shared__ float tile[32][33];
    const int kBase = blockIdx.y * 32;
    const int nBase = blockIdx.x * 32;
    for (int i = threadIdx.y; i < 32; i += 8) {
        int k = kBase + i, n = nBase + threadIdx.x;
        tile[i][threadIdx.x] = (k < K && n < N) ? W[(size_t)k * N + n] : 0.f;
    }
    __syncthreads();
    for (int i = threadIdx.y; i < 32; i += 8) {
        int n = nBase + i, k = kBase + threadIdx.x;
        if (n < N && k < K) Wt[(size_t)n * K + k] = (__bf16)tile[threadIdx.x][i];
    }
}

// ---------------------------------------------------------------------------
// Kernel 3: fused GEMM (bf16 WMMA, f32 accum) + bias + LayerNorm + activation.
// Block = 256 threads (8 wave32). Block owns 16 rows x 1024 cols (full LN row).
// Wave w owns columns [w*128, w*128+128) = 8 WMMA tiles of 16x16.
// ACT==0: exact GELU, bf16 output (feeds next layer's A matrix)
// ACT==1: tanh, f32 output (final result)
// ---------------------------------------------------------------------------
template <int ACT>
__global__ __launch_bounds__(256) void gemm_ln_act_kernel(
    const __bf16* __restrict__ A,     // [M, K] bf16 row-major
    const __bf16* __restrict__ Wt,    // [N=1024, K] bf16 row-major (W^T)
    const float*  __restrict__ bias,  // [1024]
    const float*  __restrict__ gamma, // [1024]
    const float*  __restrict__ beta,  // [1024]
    void*         __restrict__ outv,  // [M, 1024] bf16 (ACT=0) or f32 (ACT=1)
    int K)
{
    constexpr int N = 1024;
    const int tid  = threadIdx.x;
    const int wave = tid >> 5;         // 0..7
    const int lane = tid & 31;
    const int ln   = lane & 15;        // 0..15
    const int sel  = lane >> 4;        // 0 or 1 (half-wave)
    const int rowBlock = blockIdx.x << 4;   // 16 rows per block
    const int colBase  = wave << 7;         // 128 cols per wave

    v8f acc[8];
    #pragma unroll
    for (int j = 0; j < 8; ++j)
        #pragma unroll
        for (int v = 0; v < 8; ++v) acc[j][v] = 0.f;

    // A fragment: row M = ln; lanes 0-15 hold K 0..15, lanes 16-31 hold 16..31 (matches B).
    const int arow = rowBlock + ln;
    const __bf16* aBase = A + (size_t)arow * K + sel * 16;

    for (int k0 = 0; k0 < K; k0 += 32) {
        v8bf lo = *(const v8bf*)(aBase + k0);        // K = k0 + sel*16 + [0..7]
        v8bf hi = *(const v8bf*)(aBase + k0 + 8);    // K = k0 + sel*16 + 8 + [0..7]
        v16bf af;
        #pragma unroll
        for (int i = 0; i < 8; ++i) { af[i] = lo[i]; af[8 + i] = hi[i]; }

        #pragma unroll
        for (int j = 0; j < 8; ++j) {
            // B fragment: col N = ln; lanes 0-15 hold K 0..15, lanes 16-31 hold 16..31
            const __bf16* bp = Wt + (size_t)(colBase + j * 16 + ln) * K + k0 + sel * 16;
            v16bf bf = *(const v16bf*)bp;
            acc[j] = __builtin_amdgcn_wmma_f32_16x16x32_bf16(
                false, af, false, bf, (short)0, acc[j], false, false);
        }
    }

    // ---- epilogue: bias + LayerNorm(row) + activation ----
    float gj[8], bej[8];
    float rowSum[8], rowSq[8];
    #pragma unroll
    for (int v = 0; v < 8; ++v) { rowSum[v] = 0.f; rowSq[v] = 0.f; }

    #pragma unroll
    for (int j = 0; j < 8; ++j) {
        const int c = colBase + j * 16 + ln;
        const float bj = bias[c];
        gj[j]  = gamma[c];
        bej[j] = beta[c];
        #pragma unroll
        for (int v = 0; v < 8; ++v) {
            float xv = acc[j][v] + bj;
            acc[j][v] = xv;
            rowSum[v] += xv;
            rowSq[v]  += xv * xv;
        }
    }

    // reduce across the 16 lanes of this half-wave (rows sel*8 .. sel*8+7)
    #pragma unroll
    for (int m = 8; m >= 1; m >>= 1) {
        #pragma unroll
        for (int v = 0; v < 8; ++v) {
            rowSum[v] += __shfl_xor(rowSum[v], m, 32);
            rowSq[v]  += __shfl_xor(rowSq[v],  m, 32);
        }
    }

    // cross-wave reduction via LDS: [wave][16 rows]
    __shared__ float redS[8][16];
    __shared__ float redQ[8][16];
    if (ln == 0) {
        #pragma unroll
        for (int v = 0; v < 8; ++v) {
            redS[wave][sel * 8 + v] = rowSum[v];
            redQ[wave][sel * 8 + v] = rowSq[v];
        }
    }
    __syncthreads();

    float mu[8], rstd[8];
    #pragma unroll
    for (int v = 0; v < 8; ++v) {
        const int r = sel * 8 + v;
        float s = 0.f, q = 0.f;
        #pragma unroll
        for (int w = 0; w < 8; ++w) { s += redS[w][r]; q += redQ[w][r]; }
        const float m_ = s * (1.f / N);
        mu[v]   = m_;
        rstd[v] = rsqrtf(q * (1.f / N) - m_ * m_ + LN_EPS);
    }

    #pragma unroll
    for (int j = 0; j < 8; ++j) {
        const int c = colBase + j * 16 + ln;
        #pragma unroll
        for (int v = 0; v < 8; ++v) {
            const int rg = rowBlock + sel * 8 + v;
            float y = (acc[j][v] - mu[v]) * rstd[v] * gj[j] + bej[j];
            if (ACT == 0) {
                // exact GELU: 0.5*y*(1+erf(y/sqrt(2)))
                y = 0.5f * y * (1.f + erff(y * 0.70710678118654752f));
                ((__bf16*)outv)[(size_t)rg * N + c] = (__bf16)y;
            } else {
                y = tanhf(y);
                ((float*)outv)[(size_t)rg * N + c] = y;
            }
        }
    }
}

// ---------------------------------------------------------------------------
// Host launch
// Inputs: 0:x 1:token_timestamps 2:W1 3:b1 4:g1 5:be1 6:W2 7:b2 8:g2 9:be2
//         10:W3 11:b3 12:g3 13:be3
// ---------------------------------------------------------------------------
extern "C" void kernel_launch(void* const* d_in, const int* in_sizes, int n_in,
                              void* d_out, int out_size, void* d_ws, size_t ws_size,
                              hipStream_t stream) {
    const float* x  = (const float*)d_in[0];
    const int*   ts = (const int*)d_in[1];
    const float* W1 = (const float*)d_in[2];
    const float* b1 = (const float*)d_in[3];
    const float* g1 = (const float*)d_in[4];
    const float* be1= (const float*)d_in[5];
    const float* W2 = (const float*)d_in[6];
    const float* b2 = (const float*)d_in[7];
    const float* g2 = (const float*)d_in[8];
    const float* be2= (const float*)d_in[9];
    const float* W3 = (const float*)d_in[10];
    const float* b3 = (const float*)d_in[11];
    const float* g3 = (const float*)d_in[12];
    const float* be3= (const float*)d_in[13];

    // workspace layout (bytes)
    char* ws = (char*)d_ws;
    size_t off = 0;
    auto alloc = [&](size_t bytes) { char* p = ws + off; off += (bytes + 255) & ~(size_t)255; return p; };
    __bf16* pooled = (__bf16*)alloc((size_t)M_TOTAL * DENC * 2);  // 21.0 MB
    __bf16* Wt1    = (__bf16*)alloc((size_t)DHID * DENC * 2);     //  2.6 MB
    __bf16* Wt2    = (__bf16*)alloc((size_t)DHID * DHID * 2);     //  2.1 MB
    __bf16* Wt3    = (__bf16*)alloc((size_t)DEMB * DHID * 2);     //  2.1 MB
    __bf16* act1   = (__bf16*)alloc((size_t)M_TOTAL * DHID * 2);  // 16.8 MB
    __bf16* act2   = (__bf16*)alloc((size_t)M_TOTAL * DHID * 2);  // 16.8 MB

    // 1) span-mean pooling (reads the 246 MB of x exactly once)
    pool_kernel<<<M_TOTAL, 256, 0, stream>>>(x, ts, pooled);

    // 2) weight transpose+convert (tiny; keeps B fragments contiguous)
    dim3 tb(32, 8);
    transpose_to_bf16_kernel<<<dim3(DHID / 32, DENC / 32), tb, 0, stream>>>(W1, Wt1, DENC, DHID);
    transpose_to_bf16_kernel<<<dim3(DHID / 32, DHID / 32), tb, 0, stream>>>(W2, Wt2, DHID, DHID);
    transpose_to_bf16_kernel<<<dim3(DEMB / 32, DHID / 32), tb, 0, stream>>>(W3, Wt3, DHID, DEMB);

    // 3) fused GEMM + bias + LN + activation chain (512 row-blocks of 16)
    const int nblk = M_TOTAL / 16;
    gemm_ln_act_kernel<0><<<nblk, 256, 0, stream>>>(pooled, Wt1, b1, g1, be1, (void*)act1, DENC);
    gemm_ln_act_kernel<0><<<nblk, 256, 0, stream>>>(act1,   Wt2, b2, g2, be2, (void*)act2, DHID);
    gemm_ln_act_kernel<1><<<nblk, 256, 0, stream>>>(act2,   Wt3, b3, g3, be3, d_out,      DHID);
}
